// SpectralDiscriminator_21139829030979
// MI455X (gfx1250) — matrix operand order repaired and checked by reading, not compiled, not run on hardware
//
#include <hip/hip_runtime.h>
#include <hip/hip_bf16.h>
#include <math.h>

// ---------------------------------------------------------------------------
// CDNA5 (gfx1250) implementation of the spectral-norm + self-attention
// discriminator.  All conv1x1 layers and both attention blocks run through a
// generic register-blocked WMMA GEMM: each wave owns a 32x32 macro-tile
// (2x2 v_wmma_f32_16x16x32_f16 tiles, f32 accumulate) with software-pipelined
// fragment loads.
// ---------------------------------------------------------------------------

typedef _Float16 v16h __attribute__((ext_vector_type(16)));
typedef _Float16 h8   __attribute__((ext_vector_type(8)));
typedef float    v8f  __attribute__((ext_vector_type(8)));

#define LEAK 0.1f

// ---------------------------------------------------------------------------
// Spectral-norm weight preparation (one block per layer).
//   v     = l2n(W^T u)
//   sigma = u2 . (W v) = ||Wv||^2 / (||Wv|| + 1e-12)
//   Bt[o][k] = W[o][k] / sigma   (f16, zero-padded to Npad x Kpad)
// ---------------------------------------------------------------------------
__global__ void sn_prepare_kernel(const float* __restrict__ W,
                                  const float* __restrict__ u,
                                  int Cout, int Cin, int Npad, int Kpad,
                                  _Float16* __restrict__ Bt) {
  __shared__ float ush[512];
  __shared__ float vsh[512];
  __shared__ float red[256];
  const int tid = threadIdx.x;

  for (int o = tid; o < Cout; o += 256) ush[o] = u[o];
  __syncthreads();

  // t = W^T u ; accumulate ||t||^2
  float ss = 0.f;
  for (int c = tid; c < Cin; c += 256) {
    float t = 0.f;
    for (int o = 0; o < Cout; ++o) t += W[(size_t)o * Cin + c] * ush[o];
    vsh[c] = t;
    ss += t * t;
  }
  red[tid] = ss;
  __syncthreads();
  for (int s = 128; s > 0; s >>= 1) {
    if (tid < s) red[tid] += red[tid + s];
    __syncthreads();
  }
  const float vinv = 1.f / (sqrtf(red[0]) + 1e-12f);
  __syncthreads();

  for (int c = tid; c < Cin; c += 256) vsh[c] *= vinv;
  __syncthreads();

  // s = W v ; sigma = ||s||^2 / (||s|| + 1e-12)
  ss = 0.f;
  for (int o = tid; o < Cout; o += 256) {
    float t = 0.f;
    for (int c = 0; c < Cin; ++c) t += W[(size_t)o * Cin + c] * vsh[c];
    ss += t * t;
  }
  red[tid] = ss;
  __syncthreads();
  for (int s = 128; s > 0; s >>= 1) {
    if (tid < s) red[tid] += red[tid + s];
    __syncthreads();
  }
  const float ssum  = red[0];
  const float sigma = ssum / (sqrtf(ssum) + 1e-12f);
  const float isg   = 1.f / sigma;

  const int tot = Npad * Kpad;
  for (int i = tid; i < tot; i += 256) {
    const int o = i / Kpad, c = i % Kpad;
    const float w = (o < Cout && c < Cin) ? W[(size_t)o * Cin + c] * isg : 0.f;
    Bt[i] = (_Float16)w;
  }
}

// Plain f32 -> f16 weight conversion (attention q/k/v weights, padded).
__global__ void weight_f16_kernel(const float* __restrict__ W, int O, int C,
                                  int Npad, int Kpad, _Float16* __restrict__ Bt) {
  const int i = blockIdx.x * blockDim.x + threadIdx.x;
  const int tot = Npad * Kpad;
  if (i >= tot) return;
  const int o = i / Kpad, c = i % Kpad;
  Bt[i] = (_Float16)((o < O && c < C) ? W[(size_t)o * C + c] : 0.f);
}

// (B,C,H,W) f32 input  ->  pixel-major (M x 32) f16, zero-padded K (6 -> 32).
__global__ void pack_input_kernel(const float* __restrict__ x,
                                  _Float16* __restrict__ A0) {
  const int i = blockIdx.x * blockDim.x + threadIdx.x;
  if (i >= 16384 * 32) return;
  const int m = i >> 5, c = i & 31;
  const int b = m >> 12, hw = m & 4095;
  const float v = (c < 6) ? x[(size_t)b * 6 * 4096 + (size_t)c * 4096 + hw] : 0.f;
  A0[i] = (_Float16)v;
}

// ---------------------------------------------------------------------------
// Register-blocked WMMA GEMM:  D[m][n] = sum_k A[m][k] * Bt[n][k]
//   A  : M x K  f16 row-major (lda)
//   Bt : N x K  f16 row-major (ldb)   ("TN" layout)
// One 32x32 macro-tile per wave = 2x2 WMMA tiles; fragment loads for K-step
// s+1 are issued before the WMMAs of step s (software pipeline).
//
// Per-lane fragment mapping (CDNA5 ISA 7.12.2, 16-bit wave32):
//   A 16x32 : row = lane%16 ; elems j<8 -> k = half*8 + j
//                             elems j>=8 -> k = 16 + half*8 + (j-8)
//   B 32x16 : col = lane%16 ; elems j<8 / j>=8 -> same k as A
//   D 16x16 : col = lane%16 ; VGPR r    -> row = r + 8*half
// ---------------------------------------------------------------------------
#define SHUF16(lo, hi)                                                        \
  __builtin_shufflevector(lo, hi, 0, 1, 2, 3, 4, 5, 6, 7, 8, 9, 10, 11, 12,  \
                          13, 14, 15)
#define WMMA(a, b, c)                                                         \
  __builtin_amdgcn_wmma_f32_16x16x32_f16(false, a, false, b, (short)0, c,     \
                                         false, false)

__global__ void gemm_wmma_kernel(const _Float16* __restrict__ A, int lda,
                                 const _Float16* __restrict__ Bt, int ldb,
                                 const float* __restrict__ bias,
                                 const float* __restrict__ gamma_ptr,
                                 const float* __restrict__ residual,
                                 float* __restrict__ outF32,
                                 _Float16* __restrict__ outF16,
                                 _Float16* __restrict__ outF16T, int ldt,
                                 int ldc, int M, int Npad, int Nvalid, int K,
                                 int leaky) {
  const int wave = threadIdx.x >> 5;
  const int lane = threadIdx.x & 31;
  const int tile = blockIdx.x * (blockDim.x >> 5) + wave;
  const int ntn  = Npad >> 5;               // 32-wide N macro-tiles
  const int mt   = tile / ntn;
  const int nt   = tile - mt * ntn;
  if (mt >= (M >> 5)) return;               // uniform per wave

  const int m0 = mt << 5, n0 = nt << 5;
  const int hl = lane >> 4;                 // wave half (0: lanes 0-15)
  const int ln = lane & 15;

  const _Float16* a0p = A  + (size_t)(m0 + ln) * lda;
  const _Float16* a1p = a0p + (size_t)16 * lda;
  const _Float16* b0p = Bt + (size_t)(n0 + ln) * ldb;
  const _Float16* b1p = b0p + (size_t)16 * ldb;

  const int alo = hl * 8, ahi = 16 + hl * 8, bof = hl * 16;

  // Preload K-step 0 fragments.
  h8 ca0l = *(const h8*)(a0p + alo); h8 ca0h = *(const h8*)(a0p + ahi);
  h8 ca1l = *(const h8*)(a1p + alo); h8 ca1h = *(const h8*)(a1p + ahi);
  h8 cb0l = *(const h8*)(b0p + alo); h8 cb0h = *(const h8*)(b0p + ahi);
  h8 cb1l = *(const h8*)(b1p + alo); h8 cb1h = *(const h8*)(b1p + ahi);
  a0p += 32; a1p += 32; b0p += 32; b1p += 32;

  v8f acc00 = {}, acc01 = {}, acc10 = {}, acc11 = {};

  const int ksteps = K >> 5;
  for (int s = 0; s < ksteps; ++s) {
    h8 na0l = ca0l, na0h = ca0h, na1l = ca1l, na1h = ca1h;
    h8 nb0l = cb0l, nb0h = cb0h, nb1l = cb1l, nb1h = cb1h;
    if (s + 1 < ksteps) {                   // uniform branch
      na0l = *(const h8*)(a0p + alo); na0h = *(const h8*)(a0p + ahi);
      na1l = *(const h8*)(a1p + alo); na1h = *(const h8*)(a1p + ahi);
      nb0l = *(const h8*)(b0p + alo); nb0h = *(const h8*)(b0p + ahi);
      nb1l = *(const h8*)(b1p + alo); nb1h = *(const h8*)(b1p + ahi);
      __builtin_prefetch(a0p + 32, 0, 2);   // next-next tile -> L2
      __builtin_prefetch(b0p + 32, 0, 2);
      a0p += 32; a1p += 32; b0p += 32; b1p += 32;
    }
    const v16h a0 = SHUF16(ca0l, ca0h);
    const v16h a1 = SHUF16(ca1l, ca1h);
    const v16h b0 = SHUF16(cb0l, cb0h);
    const v16h b1 = SHUF16(cb1l, cb1h);
    acc00 = WMMA(a0, b0, acc00);
    acc01 = WMMA(a0, b1, acc01);
    acc10 = WMMA(a1, b0, acc10);
    acc11 = WMMA(a1, b1, acc11);
    ca0l = na0l; ca0h = na0h; ca1l = na1l; ca1h = na1h;
    cb0l = nb0l; cb0h = nb0h; cb1l = nb1l; cb1h = nb1h;
  }

  const float g = gamma_ptr ? gamma_ptr[0] : 0.f;

#pragma unroll
  for (int ni = 0; ni < 2; ++ni) {
    const int  nglob = n0 + ni * 16 + ln;
    const bool nok   = nglob < Nvalid;
    const float bv   = (bias && nok) ? bias[nglob] : 0.f;
#pragma unroll
    for (int mi = 0; mi < 2; ++mi) {
      const v8f acc = (mi == 0) ? (ni == 0 ? acc00 : acc01)
                                : (ni == 0 ? acc10 : acc11);
#pragma unroll
      for (int r = 0; r < 8; ++r) {
        const int mglob = m0 + mi * 16 + r + 8 * hl;
        float y = acc[r] + bv;
        if (leaky)    y = (y > 0.f) ? y : LEAK * y;
        if (residual) y = g * y + residual[(size_t)mglob * ldc + nglob];
        if (nok) {
          if (outF32)  outF32 [(size_t)mglob * ldc + nglob] = y;
          if (outF16)  outF16 [(size_t)mglob * ldc + nglob] = (_Float16)y;
          if (outF16T) outF16T[(size_t)nglob * ldt + mglob] = (_Float16)y;
        }
      }
    }
  }
}

// Row softmax over N f32 scores -> f16 probabilities.  One block per row.
__global__ void softmax_rows_kernel(const float* __restrict__ S,
                                    _Float16* __restrict__ P, int N) {
  __shared__ float red[256];
  const int row = blockIdx.x;
  const int tid = threadIdx.x;
  const float* __restrict__ s = S + (size_t)row * N;

  float mx = -3.4e38f;
  for (int j = tid; j < N; j += 256) mx = fmaxf(mx, s[j]);
  red[tid] = mx;
  __syncthreads();
  for (int st = 128; st > 0; st >>= 1) {
    if (tid < st) red[tid] = fmaxf(red[tid], red[tid + st]);
    __syncthreads();
  }
  mx = red[0];
  __syncthreads();

  float sum = 0.f;
  for (int j = tid; j < N; j += 256) sum += __expf(s[j] - mx);
  red[tid] = sum;
  __syncthreads();
  for (int st = 128; st > 0; st >>= 1) {
    if (tid < st) red[tid] += red[tid + st];
    __syncthreads();
  }
  const float inv = 1.f / red[0];

  _Float16* __restrict__ p = P + (size_t)row * N;
  for (int j = tid; j < N; j += 256) p[j] = (_Float16)(__expf(s[j] - mx) * inv);
}

// ---------------------------------------------------------------------------
// Host side
// ---------------------------------------------------------------------------
static void launch_gemm(hipStream_t st, const _Float16* A, int lda,
                        const _Float16* Bt, int ldb, const float* bias,
                        const float* gamma, const float* resid, float* oF,
                        _Float16* oH, _Float16* oHT, int ldt, int ldc, int M,
                        int Npad, int Nvalid, int K, int leaky) {
  const int tiles  = (M / 32) * (Npad / 32);  // 32x32 macro-tiles, div by 4
  const int blocks = tiles / 4;               // 4 waves per block
  gemm_wmma_kernel<<<blocks, 128, 0, st>>>(A, lda, Bt, ldb, bias, gamma, resid,
                                           oF, oH, oHT, ldt, ldc, M, Npad,
                                           Nvalid, K, leaky);
}

extern "C" void kernel_launch(void* const* d_in, const int* in_sizes, int n_in,
                              void* d_out, int out_size, void* d_ws,
                              size_t ws_size, hipStream_t stream) {
  (void)in_sizes; (void)n_in; (void)out_size; (void)ws_size;
  const int M = 16384;  // 4 * 64 * 64 pixels
  const int NB = 4096;  // pixels per batch

  // ---- inputs --------------------------------------------------------------
  const float* x = (const float*)d_in[0];
  const float *W[5], *bb[5], *uu[5];
  for (int i = 0; i < 5; ++i) {
    W[i]  = (const float*)d_in[1 + 3 * i + 0];
    bb[i] = (const float*)d_in[1 + 3 * i + 1];
    uu[i] = (const float*)d_in[1 + 3 * i + 2];
  }
  const float* a1_qW = (const float*)d_in[16]; const float* a1_qb = (const float*)d_in[17];
  const float* a1_kW = (const float*)d_in[18]; const float* a1_kb = (const float*)d_in[19];
  const float* a1_vW = (const float*)d_in[20]; const float* a1_vb = (const float*)d_in[21];
  const float* a1_g  = (const float*)d_in[22];
  const float* a2_qW = (const float*)d_in[23]; const float* a2_qb = (const float*)d_in[24];
  const float* a2_kW = (const float*)d_in[25]; const float* a2_kb = (const float*)d_in[26];
  const float* a2_vW = (const float*)d_in[27]; const float* a2_vb = (const float*)d_in[28];
  const float* a2_g  = (const float*)d_in[29];

  // ---- workspace bump allocator -------------------------------------------
  size_t off = 0;
  char* base = (char*)d_ws;
  auto alloc = [&](size_t bytes) -> char* {
    char* p = base + off;
    off = (off + bytes + 255) & ~(size_t)255;
    return p;
  };
  // layer dims: Cin, Cout, Kpad (mult 32), Npad (mult 32)
  const int Cin [5] = {6, 64, 128, 256, 512};
  const int Cout[5] = {64, 128, 256, 512, 1};
  const int Kpad[5] = {32, 64, 128, 256, 512};
  const int Npad[5] = {64, 128, 256, 512, 32};

  _Float16* bt[5];
  for (int i = 0; i < 5; ++i)
    bt[i] = (_Float16*)alloc((size_t)Npad[i] * Kpad[i] * 2);
  _Float16* a1qBt = (_Float16*)alloc((size_t)32 * 256 * 2);
  _Float16* a1kBt = (_Float16*)alloc((size_t)32 * 256 * 2);
  _Float16* a1vBt = (_Float16*)alloc((size_t)256 * 256 * 2);
  _Float16* a2qBt = (_Float16*)alloc((size_t)64 * 512 * 2);
  _Float16* a2kBt = (_Float16*)alloc((size_t)64 * 512 * 2);
  _Float16* a2vBt = (_Float16*)alloc((size_t)512 * 512 * 2);

  _Float16* A0      = (_Float16*)alloc((size_t)M * 32 * 2);
  _Float16* act1h   = (_Float16*)alloc((size_t)M * 64 * 2);
  _Float16* act2h   = (_Float16*)alloc((size_t)M * 128 * 2);
  float*    act3f   = (float*)   alloc((size_t)M * 256 * 4);
  _Float16* act3h   = (_Float16*)alloc((size_t)M * 256 * 2);
  _Float16* att1h   = (_Float16*)alloc((size_t)M * 256 * 2);
  float*    act4f   = (float*)   alloc((size_t)M * 512 * 4);
  _Float16* act4h   = (_Float16*)alloc((size_t)M * 512 * 2);
  _Float16* att2h   = (_Float16*)alloc((size_t)M * 512 * 2);
  _Float16* qh      = (_Float16*)alloc((size_t)M * 64 * 2);
  _Float16* kh      = (_Float16*)alloc((size_t)M * 64 * 2);
  _Float16* vT      = (_Float16*)alloc((size_t)512 * M * 2);
  float*    Sbuf    = (float*)   alloc((size_t)NB * NB * 4);
  _Float16* Pbuf    = (_Float16*)alloc((size_t)NB * NB * 2);

  // ---- weight preparation --------------------------------------------------
  pack_input_kernel<<<(M * 32 + 255) / 256, 256, 0, stream>>>(x, A0);
  for (int i = 0; i < 5; ++i)
    sn_prepare_kernel<<<1, 256, 0, stream>>>(W[i], uu[i], Cout[i], Cin[i],
                                             Npad[i], Kpad[i], bt[i]);
  auto cvt = [&](const float* w, int O, int C, _Float16* dst) {
    weight_f16_kernel<<<(O * C + 255) / 256, 256, 0, stream>>>(w, O, C, O, C, dst);
  };
  cvt(a1_qW, 32, 256, a1qBt);  cvt(a1_kW, 32, 256, a1kBt);  cvt(a1_vW, 256, 256, a1vBt);
  cvt(a2_qW, 64, 512, a2qBt);  cvt(a2_kW, 64, 512, a2kBt);  cvt(a2_vW, 512, 512, a2vBt);

  // ---- layers 1-3 ----------------------------------------------------------
  launch_gemm(stream, A0, 32, bt[0], 32, bb[0], nullptr, nullptr,
              nullptr, act1h, nullptr, 0, 64, M, 64, 64, 32, 1);
  launch_gemm(stream, act1h, 64, bt[1], 64, bb[1], nullptr, nullptr,
              nullptr, act2h, nullptr, 0, 128, M, 128, 128, 64, 1);
  launch_gemm(stream, act2h, 128, bt[2], 128, bb[2], nullptr, nullptr,
              act3f, act3h, nullptr, 0, 256, M, 256, 256, 128, 1);

  // ---- attention 1 (C=256, Ck=32) -----------------------------------------
  launch_gemm(stream, act3h, 256, a1qBt, 256, a1_qb, nullptr, nullptr,
              nullptr, qh, nullptr, 0, 32, M, 32, 32, 256, 0);
  launch_gemm(stream, act3h, 256, a1kBt, 256, a1_kb, nullptr, nullptr,
              nullptr, kh, nullptr, 0, 32, M, 32, 32, 256, 0);
  launch_gemm(stream, act3h, 256, a1vBt, 256, a1_vb, nullptr, nullptr,
              nullptr, nullptr, vT, M, 256, M, 256, 256, 256, 0);
  for (int b = 0; b < 4; ++b) {
    launch_gemm(stream, qh + (size_t)b * NB * 32, 32, kh + (size_t)b * NB * 32, 32,
                nullptr, nullptr, nullptr, Sbuf, nullptr, nullptr, 0, NB,
                NB, NB, NB, 32, 0);
    softmax_rows_kernel<<<NB, 256, 0, stream>>>(Sbuf, Pbuf, NB);
    launch_gemm(stream, Pbuf, NB, vT + (size_t)b * NB, M, nullptr, a1_g,
                act3f + (size_t)b * NB * 256, nullptr,
                att1h + (size_t)b * NB * 256, nullptr, 0, 256,
                NB, 256, 256, NB, 0);
  }

  // ---- layer 4 -------------------------------------------------------------
  launch_gemm(stream, att1h, 256, bt[3], 256, bb[3], nullptr, nullptr,
              act4f, act4h, nullptr, 0, 512, M, 512, 512, 256, 1);

  // ---- attention 2 (C=512, Ck=64) -----------------------------------------
  launch_gemm(stream, act4h, 512, a2qBt, 512, a2_qb, nullptr, nullptr,
              nullptr, qh, nullptr, 0, 64, M, 64, 64, 512, 0);
  launch_gemm(stream, act4h, 512, a2kBt, 512, a2_kb, nullptr, nullptr,
              nullptr, kh, nullptr, 0, 64, M, 64, 64, 512, 0);
  launch_gemm(stream, act4h, 512, a2vBt, 512, a2_vb, nullptr, nullptr,
              nullptr, nullptr, vT, M, 512, M, 512, 512, 512, 0);
  for (int b = 0; b < 4; ++b) {
    launch_gemm(stream, qh + (size_t)b * NB * 64, 64, kh + (size_t)b * NB * 64, 64,
                nullptr, nullptr, nullptr, Sbuf, nullptr, nullptr, 0, NB,
                NB, NB, NB, 64, 0);
    softmax_rows_kernel<<<NB, 256, 0, stream>>>(Sbuf, Pbuf, NB);
    launch_gemm(stream, Pbuf, NB, vT + (size_t)b * NB, M, nullptr, a2_g,
                act4f + (size_t)b * NB * 512, nullptr,
                att2h + (size_t)b * NB * 512, nullptr, 0, 512,
                NB, 512, 512, NB, 0);
  }

  // ---- layer 5: write directly to d_out (M floats, m-major == (B,H,W)) ----
  launch_gemm(stream, att2h, 512, bt[4], 512, bb[4], nullptr, nullptr,
              (float*)d_out, nullptr, nullptr, 0, 1, M, 32, 1, 512, 1);
}
